// Model_24077586661558
// MI455X (gfx1250) — matrix operand order repaired and checked by reading, not compiled, not run on hardware
//
#include <hip/hip_runtime.h>

// ---------------------------------------------------------------------------
// Compact Bilinear Pooling, reformulated:
//   y_b[d] = sum_{i,j} G_b[i,j] * s1[i]*s2[j] * [(h1[i]+h2[j]) & (D-1) == d]
//   G_b    = X_b X_b^T   (C x C Gram over L spatial positions)  -> bf16 WMMA
//   feat   = signsqrt(y)/max(||.||,eps);  logit = feat @ W + b
// ---------------------------------------------------------------------------

typedef __attribute__((ext_vector_type(8)))  __bf16 v8bf;
typedef __attribute__((ext_vector_type(16))) __bf16 v16bf;
typedef __attribute__((ext_vector_type(8)))  float  v8f;

#define BB    8
#define CC    2048
#define LL    784
#define LP    800      // L padded to 25 * 32 for the K=32 bf16 WMMA
#define DD    8192
#define NCLS  200

// ---------------- 1) f32 -> bf16 convert with K-padding ----------------
__global__ void cvt_bf16_kernel(const float* __restrict__ x,
                                __bf16* __restrict__ xb, int total) {
  int idx = blockIdx.x * 256 + threadIdx.x;
  if (idx >= total) return;
  int row = idx / LP;          // row = b*C + c
  int l   = idx - row * LP;
  float v = (l < LL) ? x[row * LL + l] : 0.0f;
  xb[idx] = (__bf16)v;
}

// ---------------- 2) zero the y accumulator ----------------
__global__ void zero_kernel(float* __restrict__ p, int n) {
  int idx = blockIdx.x * 256 + threadIdx.x;
  if (idx < n) p[idx] = 0.0f;
}

// ---------------- 3) W_cls [D,N] -> Wt [N,D] for coalesced dot ----------------
__global__ void trw_kernel(const float* __restrict__ W, float* __restrict__ Wt) {
  int idx = blockIdx.x * 256 + threadIdx.x;
  if (idx >= DD * NCLS) return;
  int d = idx / NCLS;
  int n = idx - d * NCLS;
  Wt[n * DD + d] = W[idx];
}

// ---------------- 4) Gram via WMMA + hash scatter into LDS bins ----------------
// grid: (C/16 row-strips, B); block: 256 threads = 8 waves.
// Each wave owns column tiles jb = wave, wave+8, ... (16 tiles); computes a
// 16x16 f32 tile of G over K=800 with v_wmma_f32_16x16x32_bf16, scatters into
// shared ybins[8192] with ds_add_f32, strip-flushes with global atomics.
__global__ __launch_bounds__(256)
void gram_scatter_kernel(const __bf16* __restrict__ xb,
                         const int*   __restrict__ h1,
                         const int*   __restrict__ h2,
                         const float* __restrict__ s1,
                         const float* __restrict__ s2,
                         float* __restrict__ y) {
  __shared__ float ybins[DD];

  const int b   = blockIdx.y;
  const int i0  = blockIdx.x * 16;
  const int tid = threadIdx.x;

  for (int d = tid; d < DD; d += 256) ybins[d] = 0.0f;
  __syncthreads();

  const int  wave = tid >> 5;
  const int  lane = tid & 31;
  const int  lmod = lane & 15;
  const bool hi   = (lane >= 16);

  const __bf16* xbb = xb + (size_t)b * CC * LP;

  // C/D VGPR layout: VGPR r holds M = r (lanes 0-15) or M = 8+r (lanes 16-31).
  const int mbase = hi ? 8 : 0;
  int   h1v[8];
  float s1v[8];
#pragma unroll
  for (int r = 0; r < 8; ++r) {
    const int i = i0 + mbase + r;
    h1v[r] = h1[i];
    s1v[r] = s1[i];
  }

  // 16-bit A layout: lane<16 -> row=lane,   K in {0..7} U {16..23}
  //                  lane>=16 -> row=lane-16, K in {8..15} U {24..31}
  const __bf16* aptr = xbb + (size_t)(i0 + lmod) * LP + (hi ? 8 : 0);

  for (int jb = wave; jb < CC / 16; jb += 8) {
    const int   j0   = jb * 16;
    const int   jcol = j0 + lmod;
    const int   h2v  = h2[jcol];
    const float s2v  = s2[jcol];
    // 16-bit B layout (B = X^T, column n = row j0+n of X):
    //   lane<16 -> col=lane,    K in {0..7} U {16..23};  lane>=16 -> col=lane-16, K in {8..15} U {24..31}
    const __bf16* bptr = xbb + (size_t)jcol * LP + (hi ? 8 : 0);

    v8f acc = {};
    for (int k = 0; k < LP; k += 32) {
      v8bf a_lo = *(const v8bf*)(aptr + k);
      v8bf a_hi = *(const v8bf*)(aptr + k + 16);
      v8bf b_lo = *(const v8bf*)(bptr + k);
      v8bf b_hi = *(const v8bf*)(bptr + k + 16);
      v16bf av = __builtin_shufflevector(a_lo, a_hi,
          0, 1, 2, 3, 4, 5, 6, 7, 8, 9, 10, 11, 12, 13, 14, 15);
      v16bf bv = __builtin_shufflevector(b_lo, b_hi,
          0, 1, 2, 3, 4, 5, 6, 7, 8, 9, 10, 11, 12, 13, 14, 15);
      acc = __builtin_amdgcn_wmma_f32_16x16x32_bf16(
          /*neg_a=*/false, av, /*neg_b=*/false, bv,
          /*c_mod=*/(short)0, acc, /*reuse_a=*/false, /*reuse_b=*/false);
    }

#pragma unroll
    for (int r = 0; r < 8; ++r) {
      const int dd = (h1v[r] + h2v) & (DD - 1);
      atomicAdd(&ybins[dd], acc[r] * s1v[r] * s2v);   // ds_add_f32
    }
  }

  __syncthreads();
  float* yb = y + (size_t)b * DD;
  for (int d = tid; d < DD; d += 256) {
    const float v = ybins[d];
    if (v != 0.0f) atomicAdd(&yb[d], v);              // global_atomic_add_f32
  }
}

// ---------------- 5) per-batch L2 norm: ||feat||^2 == sum |y| ----------------
__global__ void norms_kernel(const float* __restrict__ y, float* __restrict__ norms) {
  __shared__ float red[256];
  const int b = blockIdx.x;
  float acc = 0.0f;
  for (int d = threadIdx.x; d < DD; d += 256) acc += fabsf(y[b * DD + d]);
  red[threadIdx.x] = acc;
  __syncthreads();
  for (int s = 128; s > 0; s >>= 1) {
    if (threadIdx.x < s) red[threadIdx.x] += red[threadIdx.x + s];
    __syncthreads();
  }
  if (threadIdx.x == 0) norms[b] = fmaxf(sqrtf(red[0]), 1e-12f);
}

// ---------------- 6) signed sqrt + normalize ----------------
__global__ void finalize_kernel(const float* __restrict__ y,
                                const float* __restrict__ norms,
                                float* __restrict__ feat) {
  int idx = blockIdx.x * 256 + threadIdx.x;
  if (idx >= BB * DD) return;
  int b = idx / DD;
  float v = y[idx];
  float f = copysignf(sqrtf(fabsf(v)), v);
  feat[idx] = f / norms[b];
}

// ---------------- 7) classifier: logit[b,n] = feat[b,:] . Wt[n,:] + b[n] ----------------
__global__ void logits_kernel(const float* __restrict__ feat,
                              const float* __restrict__ Wt,
                              const float* __restrict__ bcls,
                              float* __restrict__ out) {
  __shared__ float red[256];
  const int n = blockIdx.x;
  const int b = blockIdx.y;
  const float* f = feat + (size_t)b * DD;
  const float* w = Wt + (size_t)n * DD;
  float acc = 0.0f;
  for (int d = threadIdx.x; d < DD; d += 256) acc += f[d] * w[d];
  red[threadIdx.x] = acc;
  __syncthreads();
  for (int s = 128; s > 0; s >>= 1) {
    if (threadIdx.x < s) red[threadIdx.x] += red[threadIdx.x + s];
    __syncthreads();
  }
  if (threadIdx.x == 0) out[b * NCLS + n] = red[0] + bcls[n];
}

// ---------------------------------------------------------------------------
extern "C" void kernel_launch(void* const* d_in, const int* in_sizes, int n_in,
                              void* d_out, int out_size, void* d_ws, size_t ws_size,
                              hipStream_t stream) {
  (void)in_sizes; (void)n_in; (void)out_size; (void)ws_size;

  // setup_inputs() order: x, s1, s2, W_cls, b_cls, h1, h2
  const float* x    = (const float*)d_in[0];
  const float* s1   = (const float*)d_in[1];
  const float* s2   = (const float*)d_in[2];
  const float* Wcls = (const float*)d_in[3];
  const float* bcls = (const float*)d_in[4];
  const int*   h1   = (const int*)d_in[5];
  const int*   h2   = (const int*)d_in[6];

  float* out = (float*)d_out;                 // [B*NCLS logits][B*D feat]
  float* logit_out = out;
  float* feat_out  = out + BB * NCLS;

  // workspace layout (all 256B aligned)
  char* ws = (char*)d_ws;
  __bf16* xb    = (__bf16*)(ws);                                  // 26,214,400 B
  float*  y     = (float*)(ws + 26214400);                        //    262,144 B
  float*  Wt    = (float*)(ws + 26214400 + 262144);               //  6,553,600 B
  float*  norms = (float*)(ws + 26214400 + 262144 + 6553600);     //         32 B

  // 1) convert + pad
  {
    int total = BB * CC * LP;
    cvt_bf16_kernel<<<(total + 255) / 256, 256, 0, stream>>>(x, xb, total);
  }
  // 2) zero y
  zero_kernel<<<(BB * DD + 255) / 256, 256, 0, stream>>>(y, BB * DD);
  // 3) transpose classifier weights
  {
    int total = DD * NCLS;
    trw_kernel<<<(total + 255) / 256, 256, 0, stream>>>(Wcls, Wt);
  }
  // 4) Gram (WMMA bf16) + count-sketch scatter
  {
    dim3 grid(CC / 16, BB);
    gram_scatter_kernel<<<grid, 256, 0, stream>>>(xb, h1, h2, s1, s2, y);
  }
  // 5) norms
  norms_kernel<<<BB, 256, 0, stream>>>(y, norms);
  // 6) signed sqrt + normalize -> feat output
  finalize_kernel<<<(BB * DD + 255) / 256, 256, 0, stream>>>(y, norms, feat_out);
  // 7) classifier -> logit output
  {
    dim3 grid(NCLS, BB);
    logits_kernel<<<grid, 256, 0, stream>>>(feat_out, Wt, bcls, logit_out);
  }
}
